// TFPerformerAttention_72619307041525
// MI455X (gfx1250) — matrix operand order repaired and checked by reading, not compiled, not run on hardware
//
#include <hip/hip_runtime.h>
#include <hip/hip_bf16.h>
#include <math.h>

// ---------------------------------------------------------------------------
// Performer (FAVOR+) causal linear attention for MI455X / gfx1250.
// B=2,H=16,L=4096,D=64,M=256. f16 WMMA (16x16x32), f32 accumulation.
// 3-phase chunk decomposition (256 WGs) so the whole chip participates.
// All LDS fragment traffic is contiguous b128 via fragment-major layouts.
// ---------------------------------------------------------------------------

typedef _Float16 v16h __attribute__((ext_vector_type(16)));
typedef _Float16 h8   __attribute__((ext_vector_type(8)));
typedef float    v8f  __attribute__((ext_vector_type(8)));

#define HD    64      // head dim
#define MF    256     // random features
#define DV    80      // HD + 16: col 64 = ones (denominator), 65..79 = 0
#define SUB   64      // rows per inner iteration
#define CW    512     // rows per chunk-block
#define NSUB  8       // CW / SUB
#define NCH   8       // chunks per (b,h) = L / CW
#define LSEQ  4096
#define NBH   32      // B*H

// row-major LDS strides (elements)
#define SK_STR 72     // k/q/A chunk rows (64 valid + pad), 144B rows
#define SP_STR 264    // q'/k' rows (256 valid + pad), 528B rows
#define SA_STR 72
#define SS_STR 84     // S f32 rows (80 valid + pad)

// LDS byte offsets (all 16B aligned)
#define OFF_WF  0         // W frag-major  f16: 16ft*2kk*32lane*16h   = 32768
#define OFF_K   32768     // k rows f16 64x72                         =  9216
#define OFF_Q   41984     // q rows f16 64x72                         =  9216
#define OFF_QP  51200     // q' rows f16 64x264                       = 33792
#define OFF_KP  84992     // k' rows f16 64x264                       = 33792
#define OFF_KA  118784    // k'^T A-frag-major f16 16ft*2kt*32*16     = 32768
#define OFF_A   151552    // masked attn rows f16 64x72               =  9216
#define OFF_VF  160768    // v_ext B-frag-major f16 5nd*2kt*32*16     = 10240
#define OFF_S   171008    // S master f32 256x84                      = 86016
#define OFF_SF  257024    // S shadow B-frag-major f16 8kk*5nd*32*24  = 61440
#define OFF_HK  318464    // h_k per row f32[64]                      =   256
#define OFF_DEN 318720    // denominators f32[64]                     =   256
#define LDS_BYTES 318976  // < 320KB WGP limit

#define SCALE_F  0.17677669529663687f      // 1024^-0.25
#define SCALE2_F 0.03125f                  // SCALE^2
#define CINV_M   0.0625f                   // M^-0.5
#define EPS_F    1e-4f
#define STAB_F   1e-6f

#define WMMA(a, b, c) \
    __builtin_amdgcn_wmma_f32_16x16x32_f16(false, (a), false, (b), (short)0, (c), false, false)

// ---- vectorized fragment helpers (b128 LDS traffic only) -------------------

__device__ __forceinline__ v16h cat8(h8 lo, h8 hi) {
    return __builtin_shufflevector(lo, hi, 0,1,2,3,4,5,6,7,8,9,10,11,12,13,14,15);
}
// 16 contiguous halfs -> fragment (frag-major arrays, and transposed-B reads)
__device__ __forceinline__ v16h ld_f16x16(const _Float16* p) {
    return cat8(*(const h8*)p, *(const h8*)(p + 8));
}
// A fragment (16x32 f16) from row-major tile at p0=&mem[row0][k0]
__device__ __forceinline__ v16h ld_a(const _Float16* p0, int stride, int lane) {
    int r = lane & 15, ks = (lane & 16) ? 16 : 0;
    const _Float16* p = p0 + r * stride + ks;
    return cat8(*(const h8*)p, *(const h8*)(p + 8));
}

// ---- k_stab: global max of h_k = -0.5*||k*scale||^2 (ordered-uint atomics) -

__global__ void kstab_init(unsigned int* key) { *key = 0x007FFFFFu; }  // enc(-inf)

__global__ void kstab_reduce(const float* __restrict__ Kg, unsigned int* key) {
    int row = blockIdx.x * blockDim.x + threadIdx.x;          // B*H*L rows
    const float4* p = (const float4*)(Kg + (size_t)row * HD);
    float s = 0.f;
#pragma unroll
    for (int j = 0; j < 16; ++j) {
        float4 v = p[j];
        s += v.x * v.x + v.y * v.y + v.z * v.z + v.w * v.w;
    }
    float h = -0.5f * s * SCALE2_F;
    unsigned int bits = __float_as_uint(h);
    unsigned int enc  = (bits & 0x80000000u) ? ~bits : (bits | 0x80000000u);
    atomicMax(key, enc);
}

// ---- exclusive prefix over chunk-local S sums (per (b,h), elementwise) -----

__global__ void performer_prefix(float* __restrict__ wsS) {
    const int MD = MF * DV;
    int idx = blockIdx.x * blockDim.x + threadIdx.x;
    if (idx >= NBH * MD) return;
    int bh = idx / MD, e = idx % MD;
    float* base = wsS + (size_t)bh * NCH * MD + e;
    float run = 0.f;
#pragma unroll
    for (int t = 0; t < NCH; ++t) {
        float v = base[(size_t)t * MD];
        base[(size_t)t * MD] = run;
        run += v;
    }
}

// ---- main kernel: mode 0 = accumulate chunk S sums; mode 1 = produce output

__global__ __launch_bounds__(512, 1)
void performer_main(const float* __restrict__ Qg, const float* __restrict__ Kg,
                    const float* __restrict__ Vg, const float* __restrict__ Wg,
                    float* __restrict__ Og, float* __restrict__ wsS,
                    const unsigned int* __restrict__ kstab_key, int mode)
{
    extern __shared__ char smem[];
    _Float16* sWf = (_Float16*)(smem + OFF_WF);
    _Float16* sK  = (_Float16*)(smem + OFF_K);
    _Float16* sQ  = (_Float16*)(smem + OFF_Q);
    _Float16* sQp = (_Float16*)(smem + OFF_QP);
    _Float16* sKp = (_Float16*)(smem + OFF_KP);
    _Float16* sKa = (_Float16*)(smem + OFF_KA);
    _Float16* sA  = (_Float16*)(smem + OFF_A);
    _Float16* sVf = (_Float16*)(smem + OFF_VF);
    float*    sS  = (float*)   (smem + OFF_S);
    _Float16* sSf = (_Float16*)(smem + OFF_SF);
    float*    hk  = (float*)   (smem + OFF_HK);
    float*    sDen= (float*)   (smem + OFF_DEN);

    const int tid  = threadIdx.x;
    const int lane = tid & 31;
    const int wv   = tid >> 5;            // 16 waves
    const int ch   = blockIdx.x;          // chunk within head
    const int bh   = blockIdx.y;          // (b,h)
    const int rhi  = (lane & 16) ? 8 : 0;
    const int cn   = lane & 15;

    // decode k_stab
    unsigned int kkey = *kstab_key;
    float kstab = (kkey & 0x80000000u) ? __uint_as_float(kkey ^ 0x80000000u)
                                       : __uint_as_float(~kkey);

    // stage W into B-fragment-major layout: [ (ft*2+k2)*32 + lane ][16]
    for (int i = tid; i < HD * MF; i += 512) {
        int h  = i & 15;
        int ls = (i >> 4) & 31;
        int k2 = (i >> 9) & 1;
        int ft = i >> 10;
        int k  = k2 * 32 + ((ls & 16) ? 16 : 0) + h;       // head-dim row of W
        int n  = ft * 16 + (ls & 15);                      // feature col of W
        sWf[i] = (_Float16)Wg[k * MF + n];
    }
    // init S master: zero (mode 0) or exclusive-prefix slot (mode 1)
    if (mode == 0) {
        for (int i = tid; i < MF * DV; i += 512)
            sS[(i / DV) * SS_STR + (i % DV)] = 0.f;
    } else {
        const float* src = wsS + (size_t)(bh * NCH + ch) * (MF * DV);
        for (int i = tid; i < MF * DV; i += 512)
            sS[(i / DV) * SS_STR + (i % DV)] = src[i];
    }
    __syncthreads();
    // initial f16 fragment-major shadow of S (mode 1 only; rewritten by phase F)
    if (mode) {
        for (int i = tid; i < 8 * 5 * 32 * 16; i += 512) {
            int h  = i & 15;
            int ls = (i >> 4) & 31;
            int nd = (i >> 9) % 5;
            int kk = (i >> 9) / 5;
            int k  = kk * 32 + ((ls & 16) ? 16 : 0) + h;
            int n  = nd * 16 + (ls & 15);
            sSf[((kk * 5 + nd) * 32 + ls) * 24 + h] = (_Float16)sS[k * SS_STR + n];
        }
    }
    __syncthreads();

    const size_t rowbase = ((size_t)bh * LSEQ + (size_t)ch * CW) * HD;

    for (int s = 0; s < NSUB; ++s) {
        const size_t gb = rowbase + (size_t)s * SUB * HD;

        // ---- phase A: stage k,(q) rows + v_ext frag-major; h_k via shuffles
        {
            int r = tid >> 3, c8 = tid & 7;              // 64 rows x 8 col-groups
            const float4* k4 = (const float4*)(Kg + gb + r * HD + c8 * 8);
            float part = 0.f;
            h8 kh;
#pragma unroll
            for (int j2 = 0; j2 < 2; ++j2) {
                float4 kv = k4[j2];
                kv.x *= SCALE_F; kv.y *= SCALE_F; kv.z *= SCALE_F; kv.w *= SCALE_F;
                part += kv.x * kv.x + kv.y * kv.y + kv.z * kv.z + kv.w * kv.w;
                kh[j2*4+0] = (_Float16)kv.x; kh[j2*4+1] = (_Float16)kv.y;
                kh[j2*4+2] = (_Float16)kv.z; kh[j2*4+3] = (_Float16)kv.w;
            }
            *(h8*)(sK + r * SK_STR + c8 * 8) = kh;
            part += __shfl_xor(part, 1);
            part += __shfl_xor(part, 2);
            part += __shfl_xor(part, 4);
            if ((lane & 7) == 0) hk[r] = -0.5f * part;
            // v -> fragment-major v_ext
            const float4* v4 = (const float4*)(Vg + gb + r * HD + c8 * 8);
            int kt = r >> 5;
            int lgrp = 16 * ((r >> 4) & 1);
            int p = r & 15;
            int nd = c8 >> 1;
#pragma unroll
            for (int j2 = 0; j2 < 2; ++j2) {
                float4 vv = v4[j2];
                float vs[4] = {vv.x, vv.y, vv.z, vv.w};
#pragma unroll
                for (int e = 0; e < 4; ++e) {
                    int c = c8 * 8 + j2 * 4 + e;
                    int lr = (c & 15) + lgrp;
                    sVf[((nd * 2 + kt) * 32 + lr) * 16 + p] = (_Float16)vs[e];
                }
            }
            if (c8 == 0) {   // denominator block: col 64 = 1, 65..79 = 0
#pragma unroll
                for (int e = 0; e < 16; ++e) {
                    int lr = e + lgrp;
                    sVf[((4 * 2 + kt) * 32 + lr) * 16 + p] = (_Float16)(e == 0 ? 1.f : 0.f);
                }
            }
            if (mode) {
                const float4* q4 = (const float4*)(Qg + gb + r * HD + c8 * 8);
                h8 qh;
#pragma unroll
                for (int j2 = 0; j2 < 2; ++j2) {
                    float4 qv = q4[j2];
                    qh[j2*4+0] = (_Float16)(qv.x * SCALE_F); qh[j2*4+1] = (_Float16)(qv.y * SCALE_F);
                    qh[j2*4+2] = (_Float16)(qv.z * SCALE_F); qh[j2*4+3] = (_Float16)(qv.w * SCALE_F);
                }
                *(h8*)(sQ + r * SK_STR + c8 * 8) = qh;
            }
            if (s + 1 < NSUB) {   // prefetch next sub-chunk (global_prefetch_b8)
                __builtin_prefetch(Kg + gb + SUB * HD + r * HD + c8 * 8, 0, 3);
                __builtin_prefetch(Vg + gb + SUB * HD + r * HD + c8 * 8, 0, 3);
                if (mode) __builtin_prefetch(Qg + gb + SUB * HD + r * HD + c8 * 8, 0, 3);
            }
        }
        __syncthreads();

        // ---- phase C: projections + feature maps -> sKp,sKa (and sQp) ------
        {
            int mr = wv >> 2;                 // row tile 0..3
            int f0 = (wv & 3) * 4;            // 4 feature tiles per wave
#pragma unroll
            for (int j = 0; j < 4; ++j) {
                int ft = f0 + j;
                v8f acc = {};
#pragma unroll
                for (int k2 = 0; k2 < 2; ++k2) {
                    v16h a = ld_a(sK + (mr * 16) * SK_STR + k2 * 32, SK_STR, lane);
                    v16h b = ld_f16x16(sWf + ((ft * 2 + k2) * 32 + lane) * 16);
                    acc = WMMA(a, b, acc);
                }
                h8 kv;
#pragma unroll
                for (int g = 0; g < 8; ++g) {
                    int row = mr * 16 + g + rhi;
                    float val = CINV_M * (__expf(hk[row] + acc[g] - kstab) + EPS_F);
                    kv[g] = (_Float16)val;
                    sKp[row * SP_STR + ft * 16 + cn] = kv[g];
                }
                // A-fragment-major copy of k'^T for phase F (one b128 store)
                *(h8*)(sKa + ((ft * 2 + (mr >> 1)) * 32 + cn + 16 * (mr & 1)) * 16
                           + rhi) = kv;
                if (mode) {
                    v8f acq = {};
#pragma unroll
                    for (int k2 = 0; k2 < 2; ++k2) {
                        v16h a = ld_a(sQ + (mr * 16) * SK_STR + k2 * 32, SK_STR, lane);
                        v16h b = ld_f16x16(sWf + ((ft * 2 + k2) * 32 + lane) * 16);
                        acq = WMMA(a, b, acq);
                    }
#pragma unroll
                    for (int g = 0; g < 8; ++g) {
                        int row = mr * 16 + g + rhi;
                        float val = CINV_M * (__expf(acq[g]) + EPS_F);
                        sQp[row * SP_STR + ft * 16 + cn] = (_Float16)val;
                    }
                }
            }
        }
        __syncthreads();

        // ---- phase D: intra-chunk causal attention A = tril(q' k'^T) -------
        if (mode) {
            int mr = wv >> 2, nc = wv & 3;    // one 16x16 tile per wave
            v8f acc = {};
#pragma unroll
            for (int kt = 0; kt < 8; ++kt) {
                v16h a = ld_a(sQp + (mr * 16) * SP_STR + kt * 32, SP_STR, lane);
                // B = k'^T: lane cn -> chunk col, contiguous halfs along features
                const _Float16* pb = sKp + (nc * 16 + cn) * SP_STR + kt * 32
                                   + ((lane & 16) ? 16 : 0);
                v16h b = ld_f16x16(pb);
                acc = WMMA(a, b, acc);
            }
#pragma unroll
            for (int g = 0; g < 8; ++g) {
                int i = mr * 16 + g + rhi, j = nc * 16 + cn;
                sA[i * SA_STR + j] = (_Float16)((j <= i) ? acc[g] : 0.f);
            }
        }
        __syncthreads();

        // ---- phase E: num = q' @ S_prev + A_masked @ v_ext; out = num/den --
        if (mode) {
            v8f st0 = {}, st1 = {};
            for (int t = wv; t < 20; t += 16) {          // 4 row-tiles x 5 col-tiles
                int mr = t / 5, nd = t % 5;
                v8f acc = {};
#pragma unroll
                for (int kt = 0; kt < 8; ++kt) {
                    v16h a = ld_a(sQp + (mr * 16) * SP_STR + kt * 32, SP_STR, lane);
                    v16h b = ld_f16x16(sSf + ((kt * 5 + nd) * 32 + lane) * 24);
                    acc = WMMA(a, b, acc);
                }
#pragma unroll
                for (int kt = 0; kt < 2; ++kt) {
                    v16h a = ld_a(sA + (mr * 16) * SA_STR + kt * 32, SA_STR, lane);
                    v16h b = ld_f16x16(sVf + ((nd * 2 + kt) * 32 + lane) * 16);
                    acc = WMMA(a, b, acc);
                }
                if (nd == 4 && cn == 0) {                 // publish denominators
#pragma unroll
                    for (int g = 0; g < 8; ++g) sDen[mr * 16 + g + rhi] = acc[g];
                }
                if (t < 16) st0 = acc; else st1 = acc;
            }
            __syncthreads();
            for (int t = wv; t < 20; t += 16) {
                int mr = t / 5, nd = t % 5;
                if (nd < 4) {
                    v8f acc = (t < 16) ? st0 : st1;
#pragma unroll
                    for (int g = 0; g < 8; ++g) {
                        int row = mr * 16 + g + rhi;
                        float den = sDen[row];
                        if (fabsf(den) <= STAB_F) den += 2.f * STAB_F;
                        Og[gb + (size_t)row * HD + nd * 16 + cn] = acc[g] / den;
                    }
                }
            }
        }
        __syncthreads();

        // ---- phase F: S += k'^T @ v_ext; refresh f16 fragment shadow -------
        for (int t = wv; t < 80; t += 16) {              // 16 ft x 5 nd tiles
            int ft = t / 5, nd = t % 5;
            v8f acc;
#pragma unroll
            for (int g = 0; g < 8; ++g)
                acc[g] = sS[(ft * 16 + g + rhi) * SS_STR + nd * 16 + cn];
#pragma unroll
            for (int kt = 0; kt < 2; ++kt) {
                v16h a = ld_f16x16(sKa + ((ft * 2 + kt) * 32 + lane) * 16);
                v16h b = ld_f16x16(sVf + ((nd * 2 + kt) * 32 + lane) * 16);
                acc = WMMA(a, b, acc);
            }
#pragma unroll
            for (int g = 0; g < 8; ++g)
                sS[(ft * 16 + g + rhi) * SS_STR + nd * 16 + cn] = acc[g];
            if (mode) {                                   // b128 store of shadow
                h8 sv;
#pragma unroll
                for (int g = 0; g < 8; ++g) sv[g] = (_Float16)acc[g];
                *(h8*)(sSf + (((ft >> 1) * 5 + nd) * 32 + cn + 16 * (ft & 1)) * 24
                           + rhi) = sv;
            }
        }
        __syncthreads();
    }

    // mode 0: publish chunk-local S sum to workspace
    if (mode == 0) {
        float* dst = wsS + (size_t)(bh * NCH + ch) * (MF * DV);
        for (int i = tid; i < MF * DV; i += 512)
            dst[i] = sS[(i / DV) * SS_STR + (i % DV)];
    }
}

// ---------------------------------------------------------------------------

extern "C" void kernel_launch(void* const* d_in, const int* in_sizes, int n_in,
                              void* d_out, int out_size, void* d_ws, size_t ws_size,
                              hipStream_t stream) {
    const float* Qg = (const float*)d_in[0];
    const float* Kg = (const float*)d_in[1];
    const float* Vg = (const float*)d_in[2];
    const float* Wg = (const float*)d_in[3];
    float* Og = (float*)d_out;

    unsigned int* key = (unsigned int*)d_ws;
    float* wsS = (float*)((char*)d_ws + 256);   // 32*8*256*80*4 = ~21 MB

    kstab_init<<<1, 1, 0, stream>>>(key);
    kstab_reduce<<<(NBH * LSEQ) / 256, 256, 0, stream>>>(Kg, key);

    dim3 grid(NCH, NBH);   // 8 chunks x 32 heads = 256 workgroups
    performer_main<<<grid, 512, LDS_BYTES, stream>>>(Qg, Kg, Vg, Wg, Og, wsS, key, 0);
    performer_prefix<<<(NBH * MF * DV + 255) / 256, 256, 0, stream>>>(wsS);
    performer_main<<<grid, 512, LDS_BYTES, stream>>>(Qg, Kg, Vg, Wg, Og, wsS, key, 1);
}
